// DCCF_16415365005893
// MI455X (gfx1250) — matrix operand order repaired and checked by reading, not compiled, not run on hardware
//
#include <hip/hip_runtime.h>
#include <hip/hip_bf16.h>
#include <math.h>

// ---------------------------------------------------------------------------
// Criss-cross attention (DCCF) for MI455X / gfx1250.
// All matmuls on v_wmma_f32_16x16x32_bf16 (f32 accumulate).
// bf16 intermediates staged into LDS via GLOBAL_LOAD_ASYNC_TO_LDS_B128
// (ASYNCcnt / s_wait_asynccnt) with double buffering where there is a K loop.
// B=8, C=512, CQK=64, H=W=64, HW=4096.
// ---------------------------------------------------------------------------

#define DEVI __device__ __forceinline__

static constexpr int Bn  = 8;
static constexpr int Cn  = 512;
static constexpr int CQK = 64;
static constexpr int Hn  = 64;
static constexpr int Wn  = 64;
static constexpr int HW  = Hn * Wn;     // 4096
static constexpr int LD  = 72;          // LDS row stride (bf16), multiple of 8

typedef __attribute__((ext_vector_type(16))) __bf16 bf16x16;
typedef __attribute__((ext_vector_type(8)))  float  f32x8;
typedef __attribute__((ext_vector_type(4)))  unsigned int u32x4;
typedef int v4i_vs __attribute__((vector_size(16)));   // matches builtin param type

union FragBF { bf16x16 v; u32x4 q[2]; unsigned short u[16]; };

// ---------------------------------------------------------------------------
// Async global->LDS copy path (gfx1250). Guarded: falls back to synchronous
// 16B copies if the builtins are absent on this toolchain.
// ---------------------------------------------------------------------------
#if defined(__has_builtin)
#if __has_builtin(__builtin_amdgcn_global_load_async_to_lds_b128) && \
    __has_builtin(__builtin_amdgcn_s_wait_asynccnt)
#define USE_ASYNC_LDS 1
#endif
#endif
#ifndef USE_ASYNC_LDS
#define USE_ASYNC_LDS 0
#endif

// Copy 16 bytes (8 bf16) global -> LDS. Both pointers 16B aligned.
DEVI void copy16_g2l(unsigned short* lds_dst, const unsigned short* g_src) {
#if USE_ASYNC_LDS
  __builtin_amdgcn_global_load_async_to_lds_b128(
      (__attribute__((address_space(1))) v4i_vs*)g_src,
      (__attribute__((address_space(3))) v4i_vs*)lds_dst,
      /*offset=*/0, /*cpol=*/0);
#else
  *(u32x4*)lds_dst = *(const u32x4*)g_src;
#endif
}

DEVI void async_join() {
#if USE_ASYNC_LDS
  __builtin_amdgcn_s_wait_asynccnt(0);
#endif
}

DEVI unsigned short f2bf(float f) {
  unsigned int u = __float_as_uint(f);
  unsigned int r = 0x7FFFu + ((u >> 16) & 1u);   // round-to-nearest-even
  return (unsigned short)((u + r) >> 16);
}

// A fragment (16x32 MxK, bf16) from row-major LDS tile lds[row][K], ld%8==0.
// Lanes 0-15 row M=lane; VGPR0..3 hold K=hi*8+0..7, VGPR4..7 hold
// K=16+hi*8+0..7 (hi = lane>>4).
DEVI bf16x16 load_frag_a(const unsigned short* lds, int rowBase, int kBase, int ld) {
  const int lane = threadIdx.x & 31;
  const int l = lane & 15, hi = lane >> 4;
  const unsigned short* p = lds + (rowBase + l) * ld + kBase + hi * 8;
  FragBF f;
  f.q[0] = *(const u32x4*)p;          // K = hi*8 + 0..7
  f.q[1] = *(const u32x4*)(p + 16);   // K = 16 + hi*8 + 0..7
  return f.v;
}

// B fragment (32x16 KxN, bf16) from LDS holding the tile [N][K] row-major.
// VGPR0..3 hold K=hi*8+0..7, VGPR4..7 hold K=16+hi*8+0..7 (hi = lane>>4).
DEVI bf16x16 load_frag_b(const unsigned short* lds, int nBase, int kBase, int ld) {
  const int lane = threadIdx.x & 31;
  const int l = lane & 15, hi = lane >> 4;
  const unsigned short* p = lds + (nBase + l) * ld + kBase + hi * 8;
  FragBF f;
  f.q[0] = *(const u32x4*)p;          // K = hi*8 + 0..7
  f.q[1] = *(const u32x4*)(p + 16);   // K = 16 + hi*8 + 0..7
  return f.v;
}

DEVI f32x8 wmma_bf16(bf16x16 a, bf16x16 b, f32x8 c) {
  return __builtin_amdgcn_wmma_f32_16x16x32_bf16(
      /*neg_a=*/false, a, /*neg_b=*/false, b,
      /*c_mod=*/(short)0, c, /*reuse_a=*/false, /*reuse_b=*/false);
}

// ---------------------------------------------------------------------------
// 1) 1x1 conv projections: out(M x HW) = W(M x C) * x(C x HW) + bias, per batch.
//    MODE 1: channel-last bf16 output  out[(b*HW + n)*64 + m]      (q, k)
//    MODE 0: channel-first bf16 output out[(b*M + m)*HW + n]       (v)
// fp32 inputs are converted to bf16 during LDS staging; next K-chunk is
// prefetched into L2 while the current one is computed on.
// grid: (HW/64, M/64, B), block: 128 (4 waves)
// ---------------------------------------------------------------------------
template <int MODE>
__global__ void proj_kernel(const float* __restrict__ x, const float* __restrict__ Wt,
                            const float* __restrict__ bias,
                            unsigned short* __restrict__ out, int M) {
  __shared__ __align__(16) unsigned short Ws[64 * LD];   // [m][k]
  __shared__ __align__(16) unsigned short Xs[64 * LD];   // [n][k] (transposed)
  const int t = threadIdx.x;
  const int wave = t >> 5;
  const int nBase = blockIdx.x * 64;
  const int mBase = blockIdx.y * 64;
  const int b = blockIdx.z;

  f32x8 acc[4] = {};

  for (int kBase = 0; kBase < Cn; kBase += 64) {
    for (int e = t; e < 64 * 64; e += 128) {
      int m = e >> 6, k = e & 63;
      Ws[m * LD + k] = f2bf(Wt[(size_t)(mBase + m) * Cn + kBase + k]);
    }
    for (int e = t; e < 64 * 64; e += 128) {
      int k = e >> 6, n = e & 63;   // coalesced over n
      Xs[n * LD + k] = f2bf(x[((size_t)b * Cn + kBase + k) * HW + nBase + n]);
      if (kBase + 64 < Cn)          // pull next chunk toward L2 (global_prefetch)
        __builtin_prefetch(&x[((size_t)b * Cn + kBase + 64 + k) * HW + nBase + n], 0, 0);
    }
    __syncthreads();
#pragma unroll
    for (int kc = 0; kc < 64; kc += 32) {
      bf16x16 a = load_frag_a(Ws, wave * 16, kc, LD);
#pragma unroll
      for (int nt = 0; nt < 4; ++nt) {
        bf16x16 bb = load_frag_b(Xs, nt * 16, kc, LD);
        acc[nt] = wmma_bf16(a, bb, acc[nt]);
      }
    }
    __syncthreads();
  }

  const int lane = t & 31, l = lane & 15, hi = lane >> 4;
#pragma unroll
  for (int nt = 0; nt < 4; ++nt) {
#pragma unroll
    for (int r = 0; r < 8; ++r) {
      int m = mBase + wave * 16 + r + 8 * hi;
      int n = nBase + nt * 16 + l;
      float val = acc[nt][r] + bias[m];
      if (MODE == 0) out[((size_t)b * M + m) * HW + n] = f2bf(val);
      else           out[((size_t)b * HW + n) * 64 + m] = f2bf(val);
    }
  }
}

// ---------------------------------------------------------------------------
// 2) Energies. MODEH=1: per (b,w) column slice E[i,j]=sum_c q[b,i,w,c]k[b,j,w,c],
//    diagonal masked to -inf.  MODEH=0: per (b,h) row slice.
//    Q/K tiles staged with async global->LDS copies (rows are 128B contiguous).
// grid: B*64, block: 128
// ---------------------------------------------------------------------------
template <int MODEH>
__global__ void energy_kernel(const unsigned short* __restrict__ qb,
                              const unsigned short* __restrict__ kb,
                              float* __restrict__ e_out) {
  __shared__ __align__(16) unsigned short Qs[64 * LD];   // [i][c]
  __shared__ __align__(16) unsigned short Ks[64 * LD];   // [j][c]
  const int t = threadIdx.x, wave = t >> 5;
  const int b = blockIdx.x >> 6;
  const int s = blockIdx.x & 63;          // w (MODEH) or h

  for (int e = t; e < 512; e += 128) {    // 512 x 16B segments per tile
    int r = e >> 3, seg = (e & 7) * 8;
    int n = MODEH ? (r * Wn + s) : (s * Wn + r);
    size_t base = ((size_t)b * HW + n) * 64 + seg;
    copy16_g2l(&Qs[r * LD + seg], qb + base);
    copy16_g2l(&Ks[r * LD + seg], kb + base);
  }
  async_join();
  __syncthreads();

  f32x8 acc[4] = {};
#pragma unroll
  for (int kc = 0; kc < 64; kc += 32) {
    bf16x16 a = load_frag_a(Qs, wave * 16, kc, LD);
#pragma unroll
    for (int nt = 0; nt < 4; ++nt) {
      bf16x16 bb = load_frag_b(Ks, nt * 16, kc, LD);
      acc[nt] = wmma_bf16(a, bb, acc[nt]);
    }
  }

  const int lane = t & 31, l = lane & 15, hi = lane >> 4;
#pragma unroll
  for (int nt = 0; nt < 4; ++nt) {
#pragma unroll
    for (int r = 0; r < 8; ++r) {
      int i = wave * 16 + r + 8 * hi;
      int j = nt * 16 + l;
      float val = acc[nt][r];
      if (MODEH && i == j) val = -__builtin_huge_valf();
      int n = MODEH ? (i * Wn + s) : (s * Wn + i);
      e_out[((size_t)b * HW + n) * 64 + j] = val;
    }
  }
}

// ---------------------------------------------------------------------------
// 3) Joint softmax over 128 logits per pixel (64 H + 64 W), bf16 outputs.
// One wave32 per pixel; 8 pixels per 256-thread block.
// ---------------------------------------------------------------------------
__global__ void softmax_kernel(const float* __restrict__ eH, const float* __restrict__ eW,
                               unsigned short* __restrict__ aH, unsigned short* __restrict__ aW) {
  const int t = threadIdx.x, lane = t & 31, wave = t >> 5;
  const size_t p = (size_t)blockIdx.x * 8 + wave;      // pixel id in [0, B*HW)
  const float* rH = eH + p * 64;
  const float* rW = eW + p * 64;

  float v[4];
#pragma unroll
  for (int q = 0; q < 4; ++q) {
    int g = q * 32 + lane;
    v[q] = (g < 64) ? rH[g] : rW[g - 64];
  }
  float m = fmaxf(fmaxf(v[0], v[1]), fmaxf(v[2], v[3]));
#pragma unroll
  for (int off = 16; off > 0; off >>= 1) m = fmaxf(m, __shfl_xor(m, off, 32));
  float pv[4], ssum = 0.f;
#pragma unroll
  for (int q = 0; q < 4; ++q) { pv[q] = __expf(v[q] - m); ssum += pv[q]; }
#pragma unroll
  for (int off = 16; off > 0; off >>= 1) ssum += __shfl_xor(ssum, off, 32);
  const float inv = 1.0f / ssum;
#pragma unroll
  for (int q = 0; q < 4; ++q) {
    int g = q * 32 + lane;
    unsigned short bv = f2bf(pv[q] * inv);
    if (g < 64) aH[p * 64 + g] = bv; else aW[p * 64 + g - 64] = bv;
  }
}

// ---------------------------------------------------------------------------
// 4) out_H: per (b,w): D[c,i] = sum_j v[b,c,j,w] * attH[b,i,w,j]; writes d_out.
// Attention tile async-staged; V tile is spatially strided (gather) so it is
// staged synchronously.
// grid: B*64, block: 128
// ---------------------------------------------------------------------------
__global__ void outH_kernel(const unsigned short* __restrict__ vb,
                            const unsigned short* __restrict__ aH,
                            float* __restrict__ out) {
  __shared__ __align__(16) unsigned short Vs[64 * LD];   // [c][j]
  __shared__ __align__(16) unsigned short As[64 * LD];   // [i][j]
  const int t = threadIdx.x, wave = t >> 5;
  const int b = blockIdx.x >> 6, w = blockIdx.x & 63;
  const int lane = t & 31, l = lane & 15, hi = lane >> 4;

  for (int e = t; e < 512; e += 128) {
    int i = e >> 3, seg = (e & 7) * 8;
    copy16_g2l(&As[i * LD + seg], aH + ((size_t)b * HW + i * Wn + w) * 64 + seg);
  }

  for (int cBase = 0; cBase < Cn; cBase += 64) {
    for (int e = t; e < 64 * 64; e += 128) {
      int cl = e >> 6, j = e & 63;
      Vs[cl * LD + j] = vb[((size_t)b * Cn + cBase + cl) * HW + j * Wn + w];
    }
    async_join();
    __syncthreads();

    f32x8 acc[4] = {};
#pragma unroll
    for (int kc = 0; kc < 64; kc += 32) {
      bf16x16 a = load_frag_a(Vs, wave * 16, kc, LD);
#pragma unroll
      for (int nt = 0; nt < 4; ++nt) {
        bf16x16 bb = load_frag_b(As, nt * 16, kc, LD);
        acc[nt] = wmma_bf16(a, bb, acc[nt]);
      }
    }
#pragma unroll
    for (int nt = 0; nt < 4; ++nt) {
#pragma unroll
      for (int r = 0; r < 8; ++r) {
        int c = cBase + wave * 16 + r + 8 * hi;
        int i = nt * 16 + l;
        out[((size_t)b * Cn + c) * HW + i * Wn + w] = acc[nt][r];
      }
    }
    __syncthreads();
  }
}

// ---------------------------------------------------------------------------
// 5) out_W + final: per (b,h): D[c,i] = sum_j v[b,c,h,j]*attW[b,h,i,j];
//    out = gamma*(out_H + D) + x_high.
// All tiles are 128B-row-contiguous: async staging with DOUBLE BUFFERING —
// chunk c+1 streams into LDS while WMMA runs on chunk c.
// grid: B*64, block: 128
// ---------------------------------------------------------------------------
__global__ void outW_final_kernel(const unsigned short* __restrict__ vb,
                                  const unsigned short* __restrict__ aW,
                                  const float* __restrict__ x_high,
                                  const float* __restrict__ gamma,
                                  float* __restrict__ out) {
  __shared__ __align__(16) unsigned short Vs[2][64 * LD];  // [c][j], double buffered
  __shared__ __align__(16) unsigned short As[64 * LD];     // [i][j]
  const int t = threadIdx.x, wave = t >> 5;
  const int b = blockIdx.x >> 6, h = blockIdx.x & 63;
  const int lane = t & 31, l = lane & 15, hi = lane >> 4;
  const float g = gamma[0];

  // Prologue: attention tile + first V chunk.
  for (int e = t; e < 512; e += 128) {
    int r = e >> 3, seg = (e & 7) * 8;
    copy16_g2l(&As[r * LD + seg], aW + ((size_t)b * HW + h * Wn + r) * 64 + seg);
    copy16_g2l(&Vs[0][r * LD + seg],
               vb + ((size_t)b * Cn + r) * HW + h * Wn + seg);
  }

  for (int cb = 0; cb < Cn / 64; ++cb) {
    const int cur = cb & 1;
    async_join();                 // chunk cb (and As on cb==0) resident
    __syncthreads();
    if (cb + 1 < Cn / 64) {       // stream next chunk while computing this one
      for (int e = t; e < 512; e += 128) {
        int r = e >> 3, seg = (e & 7) * 8;
        copy16_g2l(&Vs[cur ^ 1][r * LD + seg],
                   vb + ((size_t)b * Cn + (cb + 1) * 64 + r) * HW + h * Wn + seg);
      }
    }

    f32x8 acc[4] = {};
#pragma unroll
    for (int kc = 0; kc < 64; kc += 32) {
      bf16x16 a = load_frag_a(Vs[cur], wave * 16, kc, LD);
#pragma unroll
      for (int nt = 0; nt < 4; ++nt) {
        bf16x16 bb = load_frag_b(As, nt * 16, kc, LD);
        acc[nt] = wmma_bf16(a, bb, acc[nt]);
      }
    }
#pragma unroll
    for (int nt = 0; nt < 4; ++nt) {
#pragma unroll
      for (int r = 0; r < 8; ++r) {
        int c = cb * 64 + wave * 16 + r + 8 * hi;
        int i = nt * 16 + l;
        size_t idx = ((size_t)b * Cn + c) * HW + h * Wn + i;
        out[idx] = g * (out[idx] + acc[nt][r]) + x_high[idx];
      }
    }
    __syncthreads();
  }
}

// ---------------------------------------------------------------------------
extern "C" void kernel_launch(void* const* d_in, const int* in_sizes, int n_in,
                              void* d_out, int out_size, void* d_ws, size_t ws_size,
                              hipStream_t stream) {
  const float* x_low  = (const float*)d_in[0];
  const float* x_high = (const float*)d_in[1];
  const float* Wq     = (const float*)d_in[2];
  const float* bq     = (const float*)d_in[3];
  const float* Wk     = (const float*)d_in[4];
  const float* bk     = (const float*)d_in[5];
  const float* Wv     = (const float*)d_in[6];
  const float* bv     = (const float*)d_in[7];
  const float* gamma  = (const float*)d_in[8];
  float* out = (float*)d_out;

  char* ws = (char*)d_ws;
  size_t off = 0;
  auto carve = [&](size_t bytes) -> void* {
    void* p = ws + off;
    off = (off + bytes + 255) & ~(size_t)255;
    return p;
  };
  unsigned short* qb = (unsigned short*)carve((size_t)Bn * HW * CQK * 2);   // 4 MB
  unsigned short* kb = (unsigned short*)carve((size_t)Bn * HW * CQK * 2);   // 4 MB
  unsigned short* vb = (unsigned short*)carve((size_t)Bn * Cn * HW * 2);    // 32 MB
  float* eH = (float*)carve((size_t)Bn * HW * 64 * 4);                      // 8 MB
  float* eW = (float*)carve((size_t)Bn * HW * 64 * 4);                      // 8 MB
  unsigned short* aH = (unsigned short*)carve((size_t)Bn * HW * 64 * 2);    // 4 MB
  unsigned short* aW = (unsigned short*)carve((size_t)Bn * HW * 64 * 2);    // 4 MB
  (void)ws_size; (void)in_sizes; (void)n_in; (void)out_size;

  const dim3 blk(128);
  proj_kernel<1><<<dim3(HW / 64, 1, Bn), blk, 0, stream>>>(x_low,  Wq, bq, qb, CQK);
  proj_kernel<1><<<dim3(HW / 64, 1, Bn), blk, 0, stream>>>(x_high, Wk, bk, kb, CQK);
  proj_kernel<0><<<dim3(HW / 64, Cn / 64, Bn), blk, 0, stream>>>(x_high, Wv, bv, vb, Cn);
  energy_kernel<1><<<Bn * Wn, blk, 0, stream>>>(qb, kb, eH);
  energy_kernel<0><<<Bn * Hn, blk, 0, stream>>>(qb, kb, eW);
  softmax_kernel<<<Bn * HW / 8, 256, 0, stream>>>(eH, eW, aH, aW);
  outH_kernel<<<Bn * Wn, blk, 0, stream>>>(vb, aH, out);
  outW_final_kernel<<<Bn * Hn, blk, 0, stream>>>(vb, aW, x_high, gamma, out);
}
